// LinearAttention_6828998000804
// MI455X (gfx1250) — compile-verified, matrix operand order checked by reading, not run on hardware
//
#include <hip/hip_runtime.h>
#include <stdint.h>

#define HEADS 4
#define DHEAD 32
#define HH 512
#define WWIDTH 512
#define CCH 128
#define NPIX (HH * WWIDTH)        // 262144 pixels
#define DIM (HEADS * DHEAD)       // 128
#define O3 (3 * DIM)              // 384 qkv outputs per pixel

typedef __attribute__((ext_vector_type(16))) __bf16 v16bf;
typedef __attribute__((ext_vector_type(8)))  float  v8f;

union BfVec { v16bf v; uint4 u[2]; };

// ds_swizzle_b32 group-of-32 XOR patterns (and_mask=0x1F, xor_mask<<10)
#define SWZ_X1  0x041F
#define SWZ_X2  0x081F
#define SWZ_X4  0x101F
#define SWZ_X8  0x201F
#define SWZ_X16 0x401F
#define SWZI(v, p) __builtin_amdgcn_ds_swizzle((v), (p))
#define SWZF(v, p) __int_as_float(__builtin_amdgcn_ds_swizzle(__float_as_int(v), (p)))

// ---- ws layout (bytes), all offsets 256-aligned ----
#define OFF_XB   ((size_t)0)                    // x as bf16:          N*128*2 = 64 MiB
#define OFF_QB   ((size_t)67108864)             // softmaxed q, bf16:  N*128*2 = 64 MiB
#define OFF_KB   ((size_t)134217728)            // exp(k), bf16:       N*128*2 = 64 MiB
#define OFF_VB   ((size_t)201326592)            // v, bf16:            N*128*2 = 64 MiB
#define OFF_SUM  ((size_t)268435456)            // k column sumexp: 128 f32
#define OFF_CTX  ((size_t)268435968)            // raw context: 4*32*32 f32
#define OFF_WCB  ((size_t)268452352)            // folded weight bf16: 128*128
#define OFF_WIB  ((size_t)268485120)            // w_in bf16: 384*128

__device__ __forceinline__ uint16_t f2bf(float f) {
  unsigned u = __float_as_uint(f);
  unsigned r = u + 0x7FFFu + ((u >> 16) & 1u);   // round-to-nearest-even
  return (uint16_t)(r >> 16);
}

// ---- init small accumulators (re-run every call: graph-replay safe) ----
__global__ void k_init(float* __restrict__ colsum, float* __restrict__ ctx) {
  const int t = threadIdx.x;
  if (t < DIM) colsum[t] = 0.f;
  for (int i = t; i < HEADS * DHEAD * DHEAD; i += 256) ctx[i] = 0.f;
}

// ---- convert x and w_in to bf16 ----
__global__ __launch_bounds__(256) void k_cvt(const float* __restrict__ x,
                                             const float* __restrict__ w_in,
                                             uint16_t* __restrict__ xb,
                                             uint16_t* __restrict__ wib) {
  const size_t NC = (size_t)NPIX * CCH;
  const size_t TOT = NC + (size_t)O3 * CCH;
  const size_t stride = (size_t)gridDim.x * 256;
  for (size_t i = (size_t)blockIdx.x * 256 + threadIdx.x; i < TOT; i += stride) {
    if (i < NC) xb[i] = f2bf(x[i]);
    else        wib[i - NC] = f2bf(w_in[i - NC]);
  }
}

// ---- fused qkv GEMM + q-softmax + k-exp/colsum + bf16 emission.
// Block = 8 waves, 128 rows. N-tiles processed in pairs (one 32-col head slice).
// Pair loop fully unrolled (constant p folds the epilogue branches; fresh regs
// per pair let the scheduler hoist B loads). Cross-lane reductions use single
// ds_swizzle_b32 XOR patterns instead of bounds-checked ds_bpermute shuffles.
__global__ __launch_bounds__(256) void k_qkv(const uint16_t* __restrict__ xb,
                                             const uint16_t* __restrict__ wib,
                                             uint16_t* __restrict__ qb,
                                             uint16_t* __restrict__ kb,
                                             uint16_t* __restrict__ vb,
                                             float* __restrict__ colsum) {
  __shared__ float scol[DIM];
  const int t = threadIdx.x;
  if (t < DIM) scol[t] = 0.f;
  __syncthreads();

  const int wave = t >> 5;
  const int lane = t & 31;
  const int hf   = lane >> 4;
  const int r    = lane & 15;
  const int mbase = (blockIdx.x * 8 + wave) * 16;
  const bool even = ((r & 1) == 0);

  const char* aRow = (const char*)xb + (size_t)(mbase + r) * (CCH * 2);
  uint4 a0[4], a1[4];
#pragma unroll
  for (int kt = 0; kt < 4; ++kt) {
    a0[kt] = *(const uint4*)(aRow + kt * 64 + hf * 16);
    a1[kt] = *(const uint4*)(aRow + kt * 64 + 32 + hf * 16);
  }

#pragma unroll
  for (int p = 0; p < 12; ++p) {           // tile pair = cols [32p, 32p+32)
    const char* bRowA = (const char*)wib + (size_t)(2 * p * 16 + r) * (CCH * 2);
    const char* bRowB = bRowA + 16 * (CCH * 2);
    v8f accA = {0.f, 0.f, 0.f, 0.f, 0.f, 0.f, 0.f, 0.f};
    v8f accB = {0.f, 0.f, 0.f, 0.f, 0.f, 0.f, 0.f, 0.f};
#pragma unroll
    for (int kt = 0; kt < 4; ++kt) {
      BfVec A, B;
      A.u[0] = a0[kt];
      A.u[1] = a1[kt];
      B.u[0] = *(const uint4*)(bRowA + kt * 64 + hf * 16);
      B.u[1] = *(const uint4*)(bRowA + kt * 64 + 32 + hf * 16);
      accA = __builtin_amdgcn_wmma_f32_16x16x32_bf16(
          false, A.v, false, B.v, (short)0, accA, false, false);
      B.u[0] = *(const uint4*)(bRowB + kt * 64 + hf * 16);
      B.u[1] = *(const uint4*)(bRowB + kt * 64 + 32 + hf * 16);
      accB = __builtin_amdgcn_wmma_f32_16x16x32_bf16(
          false, A.v, false, B.v, (short)0, accB, false, false);
    }

    if (p < 4) {
      // ---- q: softmax over the 32 cols of head h=p, scale 1/sqrt(32) ----
#pragma unroll
      for (int i = 0; i < 8; ++i) {
        float mx = fmaxf(accA[i], accB[i]);
        mx = fmaxf(mx, SWZF(mx, SWZ_X1));
        mx = fmaxf(mx, SWZF(mx, SWZ_X2));
        mx = fmaxf(mx, SWZF(mx, SWZ_X4));
        mx = fmaxf(mx, SWZF(mx, SWZ_X8));
        float ea = __expf(accA[i] - mx);
        float eb = __expf(accB[i] - mx);
        float s = ea + eb;
        s += SWZF(s, SWZ_X1);
        s += SWZF(s, SWZ_X2);
        s += SWZF(s, SWZ_X4);
        s += SWZF(s, SWZ_X8);
        const float sc = 1.0f / (s * 5.65685424949238f);
        unsigned ba = f2bf(ea * sc), bb = f2bf(eb * sc);
        unsigned oa = (unsigned)SWZI((int)ba, SWZ_X1);
        unsigned ob = (unsigned)SWZI((int)bb, SWZ_X1);
        if (even) {
          const int row = mbase + i + 8 * hf;
          uint32_t* base = (uint32_t*)(qb + (size_t)row * DIM);
          base[(32 * p + r) >> 1]      = ba | (oa << 16);
          base[(32 * p + 16 + r) >> 1] = bb | (ob << 16);
        }
      }
    } else if (p < 8) {
      // ---- k: emit exp(k) bf16, accumulate column sum-of-exp ----
      const int cbase = 32 * (p - 4);
      float sA = 0.f, sB = 0.f;
#pragma unroll
      for (int i = 0; i < 8; ++i) {
        float ea = __expf(accA[i]);
        float eb = __expf(accB[i]);
        sA += ea; sB += eb;
        unsigned ba = f2bf(ea), bb = f2bf(eb);
        unsigned oa = (unsigned)SWZI((int)ba, SWZ_X1);
        unsigned ob = (unsigned)SWZI((int)bb, SWZ_X1);
        if (even) {
          const int row = mbase + i + 8 * hf;
          uint32_t* base = (uint32_t*)(kb + (size_t)row * DIM);
          base[(cbase + r) >> 1]      = ba | (oa << 16);
          base[(cbase + 16 + r) >> 1] = bb | (ob << 16);
        }
      }
      sA += SWZF(sA, SWZ_X16);             // merge row halves (same column set)
      sB += SWZF(sB, SWZ_X16);
      if (lane < 16) {
        atomicAdd(&scol[cbase + r], sA);
        atomicAdd(&scol[cbase + 16 + r], sB);
      }
    } else {
      // ---- v: straight bf16 emission (1/N folded into Wc later) ----
      const int cbase = 32 * (p - 8);
#pragma unroll
      for (int i = 0; i < 8; ++i) {
        unsigned ba = f2bf(accA[i]), bb = f2bf(accB[i]);
        unsigned oa = (unsigned)SWZI((int)ba, SWZ_X1);
        unsigned ob = (unsigned)SWZI((int)bb, SWZ_X1);
        if (even) {
          const int row = mbase + i + 8 * hf;
          uint32_t* base = (uint32_t*)(vb + (size_t)row * DIM);
          base[(cbase + r) >> 1]      = ba | (oa << 16);
          base[(cbase + 16 + r) >> 1] = bb | (ob << 16);
        }
      }
    }
  }

  __syncthreads();
  if (t < DIM) atomicAdd(&colsum[t], scol[t]);   // one global atomic/col/block
}

// ---- raw context[h,d,e] = sum_n expk[n,hd] * v[n,he]  (bf16 inputs, f32 acc) ----
__global__ __launch_bounds__(256) void k_context(const uint16_t* __restrict__ kb,
                                                 const uint16_t* __restrict__ vb,
                                                 float* __restrict__ ctx) {
  __shared__ float ke[8][DIM];
  __shared__ float vv[8][DIM];
  const int t = threadIdx.x;
  const int hd = t >> 1;           // 0..127 = (h,d)
  const int h = hd >> 5;
  const int eh = t & 1;            // which 16-wide half of e
  float acc[16];
#pragma unroll
  for (int e = 0; e < 16; ++e) acc[e] = 0.f;
  const size_t base = (size_t)blockIdx.x * 2048;
  for (int r0 = 0; r0 < 2048; r0 += 8) {
#pragma unroll
    for (int j = 0; j < 4; ++j) {   // stage 8 rows of expk and v, bf16 -> f32
      const int lin = t + 256 * j;  // 0..1023 u32 pairs
      const int isv = lin >> 9;
      const int row = (lin >> 6) & 7;
      const int cp = lin & 63;
      const uint16_t* src = isv ? vb : kb;
      const uint32_t w = *(const uint32_t*)(src + (base + r0 + row) * DIM + cp * 2);
      float* dstrow = isv ? vv[row] : ke[row];
      dstrow[cp * 2]     = __uint_as_float(w << 16);
      dstrow[cp * 2 + 1] = __uint_as_float(w & 0xFFFF0000u);
    }
    __syncthreads();
#pragma unroll
    for (int j = 0; j < 8; ++j) {
      const float kj = ke[j][hd];
      const float* vp = &vv[j][(h << 5) + (eh << 4)];
#pragma unroll
      for (int e = 0; e < 16; ++e) acc[e] = fmaf(kj, vp[e], acc[e]);
    }
    __syncthreads();
  }
  for (int e = 0; e < 16; ++e)
    atomicAdd(&ctx[hd * 32 + (eh << 4) + e], acc[e]);
}

// ---- fold context + w_out into one bf16 matrix Wc[c][hd] ----
__global__ __launch_bounds__(256) void k_wc(const float* __restrict__ ctx,
                                            const float* __restrict__ colsum,
                                            const float* __restrict__ w_out,
                                            uint16_t* __restrict__ wcb) {
  const int idx = blockIdx.x * 256 + threadIdx.x;  // 128*128 entries
  const int c = idx >> 7;
  const int hd = idx & 127;
  const int h = hd >> 5;
  const float scale = 1.0f / (colsum[hd] * (float)NPIX);  // softmax norm * v/N
  float s = 0.f;
#pragma unroll
  for (int e = 0; e < DHEAD; ++e)
    s += ctx[hd * 32 + e] * w_out[c * DIM + (h << 5) + e];
  wcb[c * DIM + hd] = f2bf(s * scale);
}

// ---- fused: out = q_bf16 @ Wc + b_out, then row LayerNorm. 4 waves, 64 rows ----
__global__ __launch_bounds__(128) void k_out(const uint16_t* __restrict__ qb,
                                             const uint16_t* __restrict__ wcb,
                                             const float* __restrict__ bias,
                                             const float* __restrict__ lnw,
                                             float* __restrict__ out) {
  __shared__ float st[4][16][CCH];   // 32 KiB staging for LN
  const int wave = threadIdx.x >> 5;
  const int lane = threadIdx.x & 31;
  const int hf   = lane >> 4;
  const int r    = lane & 15;
  const int mbase = (blockIdx.x * 4 + wave) * 16;

  const char* aRow = (const char*)qb + (size_t)(mbase + r) * (DIM * 2);
  uint4 a0[4], a1[4];
#pragma unroll
  for (int kt = 0; kt < 4; ++kt) {
    a0[kt] = *(const uint4*)(aRow + kt * 64 + hf * 16);
    a1[kt] = *(const uint4*)(aRow + kt * 64 + 32 + hf * 16);
  }
#pragma unroll
  for (int nt = 0; nt < 8; ++nt) {
    const char* bRow = (const char*)wcb + (size_t)(nt * 16 + r) * (DIM * 2);
    v8f acc = {0.f, 0.f, 0.f, 0.f, 0.f, 0.f, 0.f, 0.f};
#pragma unroll
    for (int kt = 0; kt < 4; ++kt) {
      BfVec A, B;
      A.u[0] = a0[kt];
      A.u[1] = a1[kt];
      B.u[0] = *(const uint4*)(bRow + kt * 64 + hf * 16);
      B.u[1] = *(const uint4*)(bRow + kt * 64 + 32 + hf * 16);
      acc = __builtin_amdgcn_wmma_f32_16x16x32_bf16(
          false, A.v, false, B.v, (short)0, acc, false, false);
    }
#pragma unroll
    for (int i = 0; i < 8; ++i)
      st[wave][i + 8 * hf][nt * 16 + r] = acc[i];
  }
  __syncthreads();

  for (int rr = 0; rr < 16; ++rr) {
    const int c0 = lane * 4;
    float4 y = *(const float4*)&st[wave][rr][c0];
    float4 b = *(const float4*)&bias[c0];
    y.x += b.x; y.y += b.y; y.z += b.z; y.w += b.w;
    float s = y.x + y.y + y.z + y.w;
    float q = y.x * y.x + y.y * y.y + y.z * y.z + y.w * y.w;
    s += SWZF(s, SWZ_X1);  q += SWZF(q, SWZ_X1);
    s += SWZF(s, SWZ_X2);  q += SWZF(q, SWZ_X2);
    s += SWZF(s, SWZ_X4);  q += SWZF(q, SWZ_X4);
    s += SWZF(s, SWZ_X8);  q += SWZF(q, SWZ_X8);
    s += SWZF(s, SWZ_X16); q += SWZF(q, SWZ_X16);
    const float mean = s * (1.0f / CCH);
    const float var  = q * (1.0f / CCH) - mean * mean;
    const float rstd = rsqrtf(var + 1e-5f);
    float4 w = *(const float4*)&lnw[c0];
    float4 o;
    o.x = (y.x - mean) * rstd * w.x;
    o.y = (y.y - mean) * rstd * w.y;
    o.z = (y.z - mean) * rstd * w.z;
    o.w = (y.w - mean) * rstd * w.w;
    *(float4*)&out[(size_t)(mbase + rr) * CCH + c0] = o;
  }
}

extern "C" void kernel_launch(void* const* d_in, const int* in_sizes, int n_in,
                              void* d_out, int out_size, void* d_ws, size_t ws_size,
                              hipStream_t stream) {
  (void)in_sizes; (void)n_in; (void)out_size; (void)ws_size;
  const float* x     = (const float*)d_in[0];
  const float* w_in  = (const float*)d_in[1];
  const float* w_out = (const float*)d_in[2];
  const float* b_out = (const float*)d_in[3];
  const float* ln_w  = (const float*)d_in[4];
  char* ws = (char*)d_ws;
  uint16_t* xb   = (uint16_t*)(ws + OFF_XB);
  uint16_t* qb   = (uint16_t*)(ws + OFF_QB);
  uint16_t* kb   = (uint16_t*)(ws + OFF_KB);
  uint16_t* vb   = (uint16_t*)(ws + OFF_VB);
  float*    csum = (float*)(ws + OFF_SUM);
  float*    ctx  = (float*)(ws + OFF_CTX);
  uint16_t* wcb  = (uint16_t*)(ws + OFF_WCB);
  uint16_t* wib  = (uint16_t*)(ws + OFF_WIB);
  float*    out  = (float*)d_out;

  k_init<<<1, 256, 0, stream>>>(csum, ctx);
  k_cvt<<<4096, 256, 0, stream>>>(x, w_in, xb, wib);
  k_qkv<<<NPIX / 128, 256, 0, stream>>>(xb, wib, qb, kb, vb, csum);
  k_context<<<NPIX / 2048, 256, 0, stream>>>(kb, vb, ctx);
  k_wc<<<64, 256, 0, stream>>>(ctx, csum, w_out, wcb);
  k_out<<<NPIX / 64, 128, 0, stream>>>(qb, wcb, b_out, ln_w, out);
}
